// AdaCL_55748675502372
// MI455X (gfx1250) — matrix operand order repaired and checked by reading, not compiled, not run on hardware
//
#include <hip/hip_runtime.h>
#include <hip/hip_bf16.h>

typedef __attribute__((ext_vector_type(16))) _Float16 v16h;
typedef __attribute__((ext_vector_type(8)))  _Float16 v8h;
typedef __attribute__((ext_vector_type(8)))  float    v8f;

#define NN 8192
#define EC 262144

#define SHUF16(a, b) __builtin_shufflevector(a, b, 0, 1, 2, 3, 4, 5, 6, 7, 8, 9, 10, 11, 12, 13, 14, 15)

// ---------------------------------------------------------------- utilities

__device__ inline void atomicMaxF(float* addr, float val) {
  unsigned int* ua = (unsigned int*)addr;
  unsigned int old = *ua;
  while (true) {
    float f = __uint_as_float(old);
    if (f >= val) break;
    unsigned int assumed = old;
    old = atomicCAS(ua, assumed, __float_as_uint(val));
    if (old == assumed) break;
  }
}

__global__ void fill_f32(float* __restrict__ p, float v, long long n) {
  long long i = (long long)blockIdx.x * blockDim.x + threadIdx.x;
  if (i < n) p[i] = v;
}

// vectorized f32 -> f16 conversion (n % 4 == 0 in all uses)
__global__ void f32_to_f16(const float* __restrict__ in, _Float16* __restrict__ out,
                           long long n) {
  long long i = 4LL * ((long long)blockIdx.x * blockDim.x + threadIdx.x);
  if (i >= n) return;
  float4 v = *(const float4*)(in + i);
  _Float16 h0 = (_Float16)v.x, h1 = (_Float16)v.y, h2 = (_Float16)v.z, h3 = (_Float16)v.w;
  _Float16* o = out + i;
  o[0] = h0; o[1] = h1; o[2] = h2; o[3] = h3;
}

// B (KxN f32 row-major) -> Bt (NxK f16 row-major)
__global__ void transpose_f16(const float* __restrict__ B, _Float16* __restrict__ Bt,
                              int K, int N) {
  int idx = blockIdx.x * blockDim.x + threadIdx.x;
  if (idx >= K * N) return;
  int k = idx / N, n2 = idx - k * N;
  Bt[(size_t)n2 * K + k] = (_Float16)B[idx];
}

// ---------------------------------------------- WMMA GEMM  C = Ah * Bt^T + bias
// Ah: MxK f16 row-major (activation), Bt: NxK f16 row-major (= B transposed).
// Full tiles only: M%64==0, N%64==0, K%32==0. No LDS, no barriers: fragments
// are loaded directly from L2-resident global memory as b128 pairs.
__global__ __launch_bounds__(128)
void wmma_gemm_tn(const _Float16* __restrict__ Ah, const _Float16* __restrict__ Bt,
                  const float* __restrict__ bias, float* __restrict__ C,
                  int N, int K, int relu) {
  int tid = threadIdx.x, wave = tid >> 5, lane = tid & 31;
  int m0 = blockIdx.x * 64, n0 = blockIdx.y * 64;
  int koff  = (lane < 16) ? 0 : 8;    // A-fragment K offset (interleaved halves)
  int koffB = (lane < 16) ? 0 : 8;    // B-fragment K offset (interleaved halves)
  const _Float16* ap  = Ah + (size_t)(m0 + wave * 16 + (lane & 15)) * K;
  const _Float16* bp0 = Bt + (size_t)(n0 + (lane & 15)) * K;
  v8f acc0 = {}, acc1 = {}, acc2 = {}, acc3 = {};

  for (int k0 = 0; k0 < K; k0 += 32) {
    v8h alo = *(const v8h*)(ap + k0 + koff);
    v8h ahi = *(const v8h*)(ap + k0 + koff + 16);
    v16h af = SHUF16(alo, ahi);
    __builtin_prefetch(ap + k0 + 64, 0, 1);
#pragma unroll
    for (int ct = 0; ct < 4; ++ct) {
      const _Float16* bp = bp0 + (size_t)(ct * 16) * K;
      v8h blo = *(const v8h*)(bp + k0 + koffB);
      v8h bhi = *(const v8h*)(bp + k0 + koffB + 16);
      v16h bf = SHUF16(blo, bhi);
      v8f& acc = ct == 0 ? acc0 : ct == 1 ? acc1 : ct == 2 ? acc2 : acc3;
      acc = __builtin_amdgcn_wmma_f32_16x16x32_f16(false, af, false, bf,
                                                   (short)0, acc, false, false);
    }
  }

  int rbase = m0 + wave * 16 + ((lane < 16) ? 0 : 8);
  int cl = lane & 15;
#pragma unroll
  for (int ct = 0; ct < 4; ++ct) {
    int col = n0 + ct * 16 + cl;
    float bsv = bias ? bias[col] : 0.f;
    v8f acc = ct == 0 ? acc0 : ct == 1 ? acc1 : ct == 2 ? acc2 : acc3;
#pragma unroll
    for (int r = 0; r < 8; ++r) {
      float v = acc[r] + bsv;
      if (relu) v = fmaxf(v, 0.f);
      C[(size_t)(rbase + r) * N + col] = v;
    }
  }
}

// tiny GEMM for N=5 / N=2 heads
__global__ void small_mm(const float* __restrict__ A, const float* __restrict__ B,
                         const float* __restrict__ bias, float* __restrict__ C,
                         int M, int Nc, int K) {
  int idx = blockIdx.x * blockDim.x + threadIdx.x;
  if (idx >= M * Nc) return;
  int i = idx / Nc, j = idx - i * Nc;
  float s = bias ? bias[j] : 0.f;
  for (int k = 0; k < K; ++k) s += A[(size_t)i * K + k] * B[(size_t)k * Nc + j];
  C[idx] = s;
}

// ----------------------------------------------------------- GAT edge kernels

__global__ void node_attn_dots(const float* __restrict__ h,
                               const float* __restrict__ asrc,
                               const float* __restrict__ adst,
                               float* __restrict__ av, float* __restrict__ dv, int n) {
  int i = blockIdx.x * blockDim.x + threadIdx.x;
  if (i >= n) return;
  const float* hi = h + (size_t)i * 64;
  float s = 0.f, d = 0.f;
#pragma unroll
  for (int k = 0; k < 64; ++k) { s += hi[k] * asrc[k]; d += hi[k] * adst[k]; }
  av[i] = s; dv[i] = d;
}

__global__ void edge_logits_max(const int* __restrict__ ei, int E_,
                                const float* __restrict__ av, const float* __restrict__ dv,
                                float* __restrict__ m, float* __restrict__ eb, int n) {
  int e = blockIdx.x * blockDim.x + threadIdx.x;
  if (e >= E_ + n) return;
  int s, d;
  if (e < E_) { s = ei[e]; d = ei[E_ + e]; } else { s = e - E_; d = s; }
  float v = av[s] + dv[d];
  v = (v >= 0.f) ? v : 0.2f * v;     // leaky_relu
  eb[e] = v;
  atomicMaxF(&m[d], v);
}

__global__ void edge_expsum(const int* __restrict__ ei, int E_,
                            const float* __restrict__ m, float* __restrict__ z,
                            float* __restrict__ eb, int n) {
  int e = blockIdx.x * blockDim.x + threadIdx.x;
  if (e >= E_ + n) return;
  int d = (e < E_) ? ei[E_ + e] : (e - E_);
  float w = __expf(eb[e] - m[d]);
  eb[e] = w;
  atomicAdd(&z[d], w);
}

__global__ void edge_aggregate(const int* __restrict__ ei, int E_,
                               const float* __restrict__ eb, const float* __restrict__ z,
                               const float* __restrict__ h, float* __restrict__ agg, int n) {
  int e = blockIdx.x;               // one block per edge, 64 feature threads
  int f = threadIdx.x;
  if (e >= E_ + n) return;
  int s, d;
  if (e < E_) { s = ei[e]; d = ei[E_ + e]; } else { s = e - E_; d = s; }
  float coef = eb[e] / (z[d] + 1e-16f);
  atomicAdd(&agg[(size_t)d * 64 + f], coef * h[(size_t)s * 64 + f]);
}

__global__ void concat_bias(const float* __restrict__ a, const float* __restrict__ ba,
                            const float* __restrict__ b, const float* __restrict__ bb,
                            float* __restrict__ out, int n, int relu) {
  int idx = blockIdx.x * blockDim.x + threadIdx.x;
  if (idx >= n * 128) return;
  int i = idx >> 7, c = idx & 127;
  float v = (c < 64) ? (a[(size_t)i * 64 + c] + ba[c])
                     : (b[(size_t)i * 64 + (c - 64)] + bb[c - 64]);
  if (relu) v = fmaxf(v, 0.f);
  out[idx] = v;
}

// ----------------------------------------------------------- batchnorm + prelu

__global__ void bn_stats(const float* __restrict__ h, float* __restrict__ mu,
                         float* __restrict__ var, int n) {
  int c = blockIdx.x;   // 256 channels
  float s = 0.f, s2 = 0.f;
  for (int i = threadIdx.x; i < n; i += blockDim.x) {
    float v = h[(size_t)i * 256 + c];
    s += v; s2 += v * v;
  }
  __shared__ float sa[256], sb[256];
  sa[threadIdx.x] = s; sb[threadIdx.x] = s2;
  __syncthreads();
  for (int o = 128; o > 0; o >>= 1) {
    if (threadIdx.x < o) { sa[threadIdx.x] += sa[threadIdx.x + o]; sb[threadIdx.x] += sb[threadIdx.x + o]; }
    __syncthreads();
  }
  if (threadIdx.x == 0) {
    float m = sa[0] / n;
    mu[c] = m;
    var[c] = sb[0] / n - m * m;
  }
}

__global__ void bn_apply_prelu(float* __restrict__ h, const float* __restrict__ mu,
                               const float* __restrict__ var, const float* __restrict__ g,
                               const float* __restrict__ b, const float* __restrict__ pa, int n) {
  int idx = blockIdx.x * blockDim.x + threadIdx.x;
  if (idx >= n * 256) return;
  int c = idx & 255;
  float v = (h[idx] - mu[c]) * rsqrtf(var[c] + 1e-5f) * g[c] + b[c];
  float a = pa[0];
  h[idx] = (v >= 0.f) ? v : a * v;
}

// ------------------------------------------------------------- l2norm -> f16

__global__ void l2norm_f16(const float* __restrict__ in, _Float16* __restrict__ oh,
                           float* __restrict__ of, int n) {
  int i = blockIdx.x * blockDim.x + threadIdx.x;
  if (i >= n) return;
  const float* v = in + (size_t)i * 64;
  float s = 0.f;
#pragma unroll
  for (int k = 0; k < 64; ++k) s += v[k] * v[k];
  float inv = 1.f / fmaxf(sqrtf(s), 1e-12f);
#pragma unroll
  for (int k = 0; k < 64; ++k) {
    float t = v[k] * inv;
    oh[(size_t)i * 64 + k] = (_Float16)t;
    if (of) of[(size_t)i * 64 + k] = t;
  }
}

// ------------------------------------------------------------- adjacency bits

__global__ void adj_set(const int* __restrict__ ni, int E_, unsigned* __restrict__ bits, int n) {
  int e = blockIdx.x * blockDim.x + threadIdx.x;
  if (e >= E_) return;
  int r = ni[e], c = ni[E_ + e];
  atomicOr(&bits[(size_t)r * (n >> 5) + (c >> 5)], 1u << (c & 31));
}

// -------------------------------------- fused  sim = sn@tn^T (+10 I)  top-8
// one block per 16-row strip; 8 waves sweep column tiles with stride 8.
__global__ __launch_bounds__(256)
void sim_topk(const _Float16* __restrict__ sn, const _Float16* __restrict__ tn,
              int n, int* __restrict__ ti) {
  __shared__ float stage[8][16][17];
  __shared__ float mv[8][16][8];
  __shared__ int   mi[8][16][8];
  int tid = threadIdx.x, wave = tid >> 5, lane = tid & 31;
  int m0 = blockIdx.x * 16;
  int koff  = (lane < 16) ? 0 : 8;
  int koffB = (lane < 16) ? 0 : 8;

  const _Float16* ap = sn + (size_t)(m0 + (lane & 15)) * 64;
  v16h a0 = SHUF16(*(const v8h*)(ap + koff),      *(const v8h*)(ap + koff + 16));
  v16h a1 = SHUF16(*(const v8h*)(ap + 32 + koff), *(const v8h*)(ap + 32 + koff + 16));

  float bv[8]; int bi[8];
#pragma unroll
  for (int k = 0; k < 8; ++k) { bv[k] = -3.4e38f; bi[k] = 0; }

  int ntiles = n >> 4;                // 512, 64 per wave -> uniform loop count
  for (int t = wave; t < ntiles; t += 8) {
    int c0 = t << 4;
    const _Float16* bp = tn + (size_t)(c0 + (lane & 15)) * 64;
    v16h b0 = SHUF16(*(const v8h*)(bp + koffB),      *(const v8h*)(bp + koffB + 16));
    v16h b1 = SHUF16(*(const v8h*)(bp + 32 + koffB), *(const v8h*)(bp + 32 + koffB + 16));

    v8f acc = {};
    acc = __builtin_amdgcn_wmma_f32_16x16x32_f16(false, a0, false, b0, (short)0, acc, false, false);
    acc = __builtin_amdgcn_wmma_f32_16x16x32_f16(false, a1, false, b1, (short)0, acc, false, false);

    int cl = lane & 15, rb = (lane < 16) ? 0 : 8;
#pragma unroll
    for (int r = 0; r < 8; ++r) stage[wave][rb + r][cl] = acc[r];
    __syncthreads();

    if (lane < 16) {
      int row = lane;
#pragma unroll
      for (int c = 0; c < 16; ++c) {
        float v = stage[wave][row][c];
        int col = c0 + c;
        if (col == m0 + row) v += 10.f;          // diagonal boost
        if (v > bv[7]) {
          bv[7] = v; bi[7] = col;
#pragma unroll
          for (int j = 7; j > 0; --j) {
            if (bv[j] > bv[j - 1]) {
              float tv = bv[j]; bv[j] = bv[j - 1]; bv[j - 1] = tv;
              int tii = bi[j]; bi[j] = bi[j - 1]; bi[j - 1] = tii;
            }
          }
        }
      }
    }
    __syncthreads();
  }

  if (lane < 16) {
#pragma unroll
    for (int k = 0; k < 8; ++k) { mv[wave][lane][k] = bv[k]; mi[wave][lane][k] = bi[k]; }
  }
  __syncthreads();

  if (wave == 0 && lane < 16) {
    float fv[8]; int fi[8];
#pragma unroll
    for (int k = 0; k < 8; ++k) { fv[k] = -3.4e38f; fi[k] = 0; }
    for (int w = 0; w < 8; ++w) {
#pragma unroll
      for (int k = 0; k < 8; ++k) {
        float v = mv[w][lane][k]; int c = mi[w][lane][k];
        if (v > fv[7]) {
          fv[7] = v; fi[7] = c;
#pragma unroll
          for (int j = 7; j > 0; --j) {
            if (fv[j] > fv[j - 1]) {
              float tv = fv[j]; fv[j] = fv[j - 1]; fv[j - 1] = tv;
              int tii = fi[j]; fi[j] = fi[j - 1]; fi[j - 1] = tii;
            }
          }
        }
      }
    }
#pragma unroll
    for (int k = 0; k < 8; ++k) ti[(size_t)(m0 + lane) * 8 + k] = fi[k];
  }
}

// ------------------------------------------------------------------ k-means

__global__ void kmeans_init(const float* __restrict__ f, float* __restrict__ cent,
                            int seed, int n) {
  int j = threadIdx.x;
  if (j >= 32) return;
  int idx = ((seed + 1) * 1973 + j * 613 + 7) % n;
#pragma unroll
  for (int k = 0; k < 64; ++k) cent[j * 64 + k] = f[(size_t)idx * 64 + k];
}

__global__ __launch_bounds__(128)
void kmeans_assign(const float* __restrict__ f, const float* __restrict__ cent,
                   float* __restrict__ sums, float* __restrict__ cnt,
                   int* __restrict__ lbl, int n) {
  __shared__ float cs[32 * 64];
  for (int t = threadIdx.x; t < 32 * 64; t += 128) cs[t] = cent[t];
  __syncthreads();
  int i = blockIdx.x * blockDim.x + threadIdx.x;
  if (i >= n) return;
  const float* fi = f + (size_t)i * 64;
  float best = 3.4e38f; int bj = 0;
  for (int j = 0; j < 32; ++j) {
    const float* c = cs + j * 64;
    float s = 0.f;
#pragma unroll
    for (int k = 0; k < 64; ++k) { float d = fi[k] - c[k]; s += d * d; }
    if (s < best) { best = s; bj = j; }
  }
  lbl[i] = bj;
  if (sums) {
#pragma unroll
    for (int k = 0; k < 64; ++k) atomicAdd(&sums[bj * 64 + k], fi[k]);
    atomicAdd(&cnt[bj], 1.f);
  }
}

__global__ void kmeans_update(float* __restrict__ cent, const float* __restrict__ sums,
                              const float* __restrict__ cnt) {
  int idx = blockIdx.x * blockDim.x + threadIdx.x;
  if (idx >= 32 * 64) return;
  float c = cnt[idx >> 6];
  if (c > 0.f) cent[idx] = sums[idx] / c;
}

// ---------------------------------------------------------------- final mask

__global__ void finalize_knn(const int* __restrict__ ti, const unsigned* __restrict__ bits,
                             const int* __restrict__ lbl, float* __restrict__ o_idx,
                             float* __restrict__ o_mask, int n) {
  int idx = blockIdx.x * blockDim.x + threadIdx.x;
  if (idx >= n * 8) return;
  int i = idx >> 3;
  int j = ti[idx];
  unsigned b = bits[(size_t)i * (n >> 5) + (j >> 5)];
  bool loc = (b >> (j & 31)) & 1u;
  bool g = false;
  for (int s = 0; s < 3; ++s) g = g || (lbl[s * n + i] == lbl[s * n + j]);
  o_idx[idx]  = (float)j;
  o_mask[idx] = (loc || g) ? 1.f : 0.f;
}

// =================================================================== driver

static void gemm(const float* A, const float* B, const float* bias, float* C,
                 int M, int N, int K, int relu,
                 _Float16* ah, _Float16* bt, hipStream_t st) {
  f32_to_f16<<<((long long)M * K / 4 + 255) / 256, 256, 0, st>>>(A, ah, (long long)M * K);
  transpose_f16<<<((long long)K * N + 255) / 256, 256, 0, st>>>(B, bt, K, N);
  wmma_gemm_tn<<<dim3(M / 64, N / 64), 128, 0, st>>>(ah, bt, bias, C, N, K, relu);
}

static void run_conv(const float* xin, int K, const int* ei,
                     const float* W, const float* asrc, const float* adst,
                     float* h, float* agg,
                     float* av, float* dv, float* m, float* z, float* eb,
                     _Float16* ah, _Float16* bt, hipStream_t st) {
  gemm(xin, W, nullptr, h, NN, 64, K, 0, ah, bt, st);
  node_attn_dots<<<(NN + 255) / 256, 256, 0, st>>>(h, asrc, adst, av, dv, NN);
  fill_f32<<<(NN + 255) / 256, 256, 0, st>>>(m, -3.0e38f, NN);
  fill_f32<<<(NN + 255) / 256, 256, 0, st>>>(z, 0.f, NN);
  fill_f32<<<((long long)NN * 64 + 255) / 256, 256, 0, st>>>(agg, 0.f, (long long)NN * 64);
  int tot = EC + NN;
  edge_logits_max<<<(tot + 255) / 256, 256, 0, st>>>(ei, EC, av, dv, m, eb, NN);
  edge_expsum<<<(tot + 255) / 256, 256, 0, st>>>(ei, EC, m, z, eb, NN);
  edge_aggregate<<<tot, 64, 0, st>>>(ei, EC, eb, z, h, agg, NN);
}

extern "C" void kernel_launch(void* const* d_in, const int* in_sizes, int n_in,
                              void* d_out, int out_size, void* d_ws, size_t ws_size,
                              hipStream_t stream) {
  (void)in_sizes; (void)n_in; (void)out_size; (void)ws_size;

  const float* x   = (const float*)d_in[0];
  const int* ei_o  = (const int*)d_in[1];
  const int* ei_s  = (const int*)d_in[2];
  const int* ni    = (const int*)d_in[3];
  // params_student: o1{W,a_src,a_dst,b} s1{...} o2{...} s2{...} Wp bp = d_in[4..21]
  // params_teacher: d_in[22..39]; params_head: d_in[40..54]
  const float* PS[18]; const float* PT[18];
  for (int k = 0; k < 18; ++k) { PS[k] = (const float*)d_in[4 + k]; PT[k] = (const float*)d_in[22 + k]; }
  const float* W1  = (const float*)d_in[40]; const float* b1  = (const float*)d_in[41];
  const float* bng = (const float*)d_in[42]; const float* bnb = (const float*)d_in[43];
  const float* pa  = (const float*)d_in[44];
  const float* W2  = (const float*)d_in[45]; const float* b2  = (const float*)d_in[46];
  const float* Wc  = (const float*)d_in[47]; const float* bc  = (const float*)d_in[48];
  const float* Wd1 = (const float*)d_in[49]; const float* bd1 = (const float*)d_in[50];
  const float* Wd2 = (const float*)d_in[51]; const float* bd2 = (const float*)d_in[52];
  const float* Wd3 = (const float*)d_in[53]; const float* bd3 = (const float*)d_in[54];

  // ---- workspace arena
  char* cur = (char*)d_ws;
  auto alloc = [&](size_t b) -> char* { char* r = cur; cur += (b + 255) & ~(size_t)255; return r; };
  float* h_buf = (float*)alloc((size_t)NN * 64 * 4);
  float* agg_a = (float*)alloc((size_t)NN * 64 * 4);
  float* agg_b = (float*)alloc((size_t)NN * 64 * 4);
  float* x1    = (float*)alloc((size_t)NN * 128 * 4);
  float* x2    = (float*)alloc((size_t)NN * 128 * 4);
  float* av    = (float*)alloc((size_t)NN * 4);
  float* dv    = (float*)alloc((size_t)NN * 4);
  float* mbuf  = (float*)alloc((size_t)NN * 4);
  float* zbuf  = (float*)alloc((size_t)NN * 4);
  float* ebuf  = (float*)alloc((size_t)(EC + NN) * 4);
  float* h1    = (float*)alloc((size_t)NN * 256 * 4);
  float* dbuf1 = (float*)alloc((size_t)NN * 128 * 4);
  float* dbuf2 = (float*)alloc((size_t)NN * 128 * 4);
  float* tn_f  = (float*)alloc((size_t)NN * 64 * 4);
  float* cent  = (float*)alloc((size_t)32 * 64 * 4);
  float* sums  = (float*)alloc((size_t)(32 * 64 + 32) * 4);
  float* cnt   = sums + 32 * 64;
  int*   lbl   = (int*)alloc((size_t)3 * NN * 4);
  int*   ti    = (int*)alloc((size_t)NN * 8 * 4);
  unsigned* adjb = (unsigned*)alloc((size_t)NN * (NN / 32) * 4);
  _Float16* sn_h = (_Float16*)alloc((size_t)NN * 64 * 2);
  _Float16* tn_h = (_Float16*)alloc((size_t)NN * 64 * 2);
  _Float16* ahb  = (_Float16*)alloc((size_t)NN * 256 * 2);   // activation f16 staging
  _Float16* btb  = (_Float16*)alloc((size_t)256 * 256 * 2);  // weight^T f16 staging

  // ---- output regions: student | teacher | projection | I_knn | pos_mask | pred | dlog
  float* out       = (float*)d_out;
  float* o_student = out;
  float* o_teacher = out + (size_t)NN * 64;
  float* o_proj    = out + (size_t)NN * 64 * 2;
  float* o_iknn    = out + (size_t)NN * 64 * 3;
  float* o_mask    = o_iknn + (size_t)NN * 8;
  float* o_pred    = o_mask + (size_t)NN * 8;
  float* o_dlog    = o_pred + (size_t)NN * 5;

  hipStream_t st = stream;

  // ---- encoders (student, teacher)
  auto run_encoder = [&](const float* const* P, const int* eo, const int* es, float* emb) {
    run_conv(x, 128, eo, P[0],  P[1],  P[2],  h_buf, agg_a, av, dv, mbuf, zbuf, ebuf, ahb, btb, st);
    run_conv(x, 128, es, P[4],  P[5],  P[6],  h_buf, agg_b, av, dv, mbuf, zbuf, ebuf, ahb, btb, st);
    concat_bias<<<((long long)NN * 128 + 255) / 256, 256, 0, st>>>(agg_a, P[3], agg_b, P[7], x1, NN, 1);
    run_conv(x1, 128, eo, P[8],  P[9],  P[10], h_buf, agg_a, av, dv, mbuf, zbuf, ebuf, ahb, btb, st);
    run_conv(x1, 128, es, P[12], P[13], P[14], h_buf, agg_b, av, dv, mbuf, zbuf, ebuf, ahb, btb, st);
    concat_bias<<<((long long)NN * 128 + 255) / 256, 256, 0, st>>>(agg_a, P[11], agg_b, P[15], x2, NN, 0);
    gemm(x2, P[16], P[17], emb, NN, 64, 128, 0, ahb, btb, st);
  };
  run_encoder(PS, ei_o, ei_s, o_student);
  run_encoder(PT, ei_o, ei_s, o_teacher);

  // ---- head
  gemm(o_student, W1, b1, h1, NN, 256, 64, 0, ahb, btb, st);
  bn_stats<<<256, 256, 0, st>>>(h1, cent /*reuse as mu*/, cent + 256 /*var*/, NN);
  bn_apply_prelu<<<((long long)NN * 256 + 255) / 256, 256, 0, st>>>(h1, cent, cent + 256, bng, bnb, pa, NN);
  gemm(h1, W2, b2, o_proj, NN, 64, 256, 0, ahb, btb, st);
  small_mm<<<((long long)NN * 5 + 255) / 256, 256, 0, st>>>(o_student, Wc, bc, o_pred, NN, 5, 64);
  gemm(o_student, Wd1, bd1, dbuf1, NN, 128, 64, 1, ahb, btb, st);
  gemm(dbuf1, Wd2, bd2, dbuf2, NN, 128, 128, 1, ahb, btb, st);
  small_mm<<<((long long)NN * 2 + 255) / 256, 256, 0, st>>>(dbuf2, Wd3, bd3, o_dlog, NN, 2, 128);

  // ---- normalize embeddings (f16 for WMMA; f32 teacher copy for k-means)
  l2norm_f16<<<(NN + 255) / 256, 256, 0, st>>>(o_student, sn_h, nullptr, NN);
  l2norm_f16<<<(NN + 255) / 256, 256, 0, st>>>(o_teacher, tn_h, tn_f, NN);

  // ---- dense adjacency as bitset (8 MB instead of 64 MB bool)
  hipMemsetAsync(adjb, 0, (size_t)NN * (NN / 32) * 4, st);
  adj_set<<<(EC + 255) / 256, 256, 0, st>>>(ni, EC, adjb, NN);

  // ---- fused similarity + top-8 (WMMA, tn resident in L2)
  sim_topk<<<NN / 16, 256, 0, st>>>(sn_h, tn_h, NN, ti);

  // ---- 3x k-means on normalized teacher features
  for (int s = 0; s < 3; ++s) {
    kmeans_init<<<1, 32, 0, st>>>(tn_f, cent, s, NN);
    for (int it = 0; it < 5; ++it) {
      fill_f32<<<(32 * 64 + 32 + 255) / 256, 256, 0, st>>>(sums, 0.f, 32 * 64 + 32);
      kmeans_assign<<<NN / 128, 128, 0, st>>>(tn_f, cent, sums, cnt, lbl + s * NN, NN);
      kmeans_update<<<(32 * 64 + 63) / 64, 64, 0, st>>>(cent, sums, cnt);
    }
    kmeans_assign<<<NN / 128, 128, 0, st>>>(tn_f, cent, nullptr, nullptr, lbl + s * NN, NN);
  }

  // ---- locality | global mask + I_knn as float outputs
  finalize_knn<<<((long long)NN * 8 + 255) / 256, 256, 0, st>>>(ti, adjb, lbl, o_iknn, o_mask, NN);
}
